// Hydra_82222853914822
// MI455X (gfx1250) — matrix operand order repaired and checked by reading, not compiled, not run on hardware
//
#include <hip/hip_runtime.h>
#include <hip/hip_bf16.h>
#include <math.h>

// ---------------------------------------------------------------------------
// WaveNet stack on gfx1250 (MI455X). All channel mixing = bf16 WMMA with f32
// accumulation. Activations are TIME-MAJOR ([B][T][C]) so WMMA B-fragments are
// two ds_load_b128 per lane and global<->LDS copies are b128 transfers.
// Weights are pre-repacked into WMMA A-fragment order (bf16).
// Activation windows are pulled into LDS by the Tensor Data Mover (6-arg
// builtin on this toolchain), completion via s_wait_tensorcnt.
// ---------------------------------------------------------------------------

typedef __attribute__((ext_vector_type(16))) __bf16 v16bf;
typedef __attribute__((ext_vector_type(8)))  __bf16 v8bf;
typedef __attribute__((ext_vector_type(8)))  float  v8f;
typedef __attribute__((ext_vector_type(4)))  unsigned int u32x4;
typedef __attribute__((ext_vector_type(8)))  int   i32x8;
typedef __attribute__((ext_vector_type(4)))  int   i32x4;

#define WN_B   4
#define WN_T   8192
#define WN_R   128      // residual width
#define WN_S   64       // skip width
#define WN_L   10
#define TT     128      // time tile per workgroup
#define CPITCH 136      // LDS pitch (halves) for 128-ch tiles: 272B rows, 16B aligned
#define SPITCH 72       // LDS pitch for 64-ch tiles
#define ZPITCH 40       // LDS pitch for 32-ch tiles
#define NTILE  (WN_T / TT)

#if defined(__HIP_DEVICE_COMPILE__) && \
    __has_builtin(__builtin_amdgcn_tensor_load_to_lds) && \
    __has_builtin(__builtin_amdgcn_s_wait_tensorcnt)
#define WN_TDM 1
#else
#define WN_TDM 0
#endif

static __device__ __forceinline__ v8f wmma_bf16(v16bf a, v16bf b, v8f c) {
  return __builtin_amdgcn_wmma_f32_16x16x32_bf16(false, a, false, b, (short)0, c,
                                                 false, false);
}

// A-fragment from pre-packed weights: lane's 16 halves are contiguous (32B).
static __device__ __forceinline__ v16bf load_a_packed(const __bf16* __restrict__ P,
                                                      int tile) {
  const int lane = threadIdx.x & 31;
  const __bf16* p = P + ((long)tile * 32 + lane) * 16;
  const v8bf lo = *(const v8bf*)p;
  const v8bf hi = *(const v8bf*)(p + 8);
  return __builtin_shufflevector(lo, hi, 0, 1, 2, 3, 4, 5, 6, 7,
                                 8, 9, 10, 11, 12, 13, 14, 15);
}

// B-fragment from a time-major LDS tile sh[t][pitch]: lanes 0-15 -> K=k0+e,
// lanes 16-31 -> K=k0+16+e, N = n0 + (lane&15). Two aligned ds_load_b128.
static __device__ __forceinline__ v16bf load_b_lds(const __bf16* sh, int pitch,
                                                   int k0, int n0) {
  const int lane = threadIdx.x & 31;
  const int n  = n0 + (lane & 15);
  const int kb = k0 + ((lane >> 4) << 4);
  const __bf16* p = sh + (size_t)n * pitch + kb;
  const v8bf lo = *(const v8bf*)p;
  const v8bf hi = *(const v8bf*)(p + 8);
  return __builtin_shufflevector(lo, hi, 0, 1, 2, 3, 4, 5, 6, 7,
                                 8, 9, 10, 11, 12, 13, 14, 15);
}

static __device__ __forceinline__ v8bf zero_v8bf() {
  v8bf z;
#pragma unroll
  for (int q = 0; q < 8; ++q) z[q] = (__bf16)0.0f;
  return z;
}

// ---------------------------------------------------------------------------
// Repack an f32 weight matrix into WMMA A-fragment order as bf16.
// out[mat][tile = mb*nks + ks][lane][e] with K=kb+e,
// kb = (lane>>4)*16, M = mb*16 + (lane&15).
// ---------------------------------------------------------------------------
__global__ void wn_repack_a(const float* __restrict__ W, __bf16* __restrict__ out,
                            int nmat, long in_stride, long out_stride,
                            int nmb, int nks, int row_stride, int col_stride)
{
  const long per_mat = (long)nmb * nks * 512;
  const long i = (long)blockIdx.x * blockDim.x + threadIdx.x;
  if (i >= (long)nmat * per_mat) return;
  const int  mat  = (int)(i / per_mat);
  const long r    = i % per_mat;
  const int  tile = (int)(r >> 9);
  const int  l    = (int)((r >> 4) & 31);
  const int  e    = (int)(r & 15);
  const int  mb = tile / nks, ks = tile % nks;
  const int  m  = mb * 16 + (l & 15);
  const int  kb = (l >> 4) << 4;
  const int  k  = ks * 32 + kb + e;
  out[mat * out_stride + ((long)tile * 32 + l) * 16 + e] =
      (__bf16)W[mat * in_stride + (long)m * row_stride + (long)k * col_stride];
}

// ---------------------------------------------------------------------------
// Gated residual layer (time-major h): one (b, 128t) tile per workgroup.
// ---------------------------------------------------------------------------
__global__ __launch_bounds__(256, 1) void wn_layer(
    const __bf16* __restrict__ h_in,   // [B][T][R]
    __bf16* __restrict__ h_out,        // [B][T][R]
    float* __restrict__ skip_acc,      // [B][T][S]
    const __bf16* __restrict__ pdil,   // packed, [tap][64 tiles][512]
    const float* __restrict__ dil_b,
    const __bf16* __restrict__ pres,   // packed, [32 tiles][512]
    const float* __restrict__ res_b,
    const __bf16* __restrict__ pskip,  // packed, [16 tiles][512]
    const float* __restrict__ skip_b,
    int dilation, int first)
{
  __shared__ __align__(32) __bf16 sh[TT * CPITCH];   // window, then z (reused)

  const int tid  = threadIdx.x;
  const int wave = tid >> 5;
  const int lane = tid & 31;
  const int b    = blockIdx.x / NTILE;
  const int t0   = (blockIdx.x % NTILE) * TT;
  const int mrow = (lane >> 4) << 3;
  const int col0 = lane & 15;

  const __bf16* hin_b = h_in + (size_t)b * WN_T * WN_R;

  v8f accT[8], accS[8];
#pragma unroll
  for (int nb = 0; nb < 8; ++nb)
#pragma unroll
    for (int r = 0; r < 8; ++r) {
      accT[nb][r] = dil_b[wave * 16 + mrow + r];
      accS[nb][r] = dil_b[WN_R + wave * 16 + mrow + r];
    }

  for (int kk = 0; kk < 3; ++kk) {          // tap kk reads h[t - (2-kk)*d]
    const int shift  = (2 - kk) * dilation;
    const int tstart = t0 - shift;
    int skiprows = 0;
    if (tstart < 0) skiprows = (-tstart < TT) ? -tstart : TT;

    __syncthreads();
#if WN_TDM
    // zero causal rows, then wave0 drives the Tensor Data Mover for the rest
    for (int idx = tid; idx < skiprows * 16; idx += 256) {
      const int j = idx >> 4, c8 = (idx & 15) << 3;
      *(v8bf*)(sh + j * CPITCH + c8) = zero_v8bf();
    }
    if (wave == 0 && skiprows < TT) {
      const unsigned long long gaddr =
          (unsigned long long)(const void*)(hin_b + (size_t)(tstart + skiprows) * WN_R);
      const unsigned lds_addr =
          (unsigned)(size_t)(sh) + (unsigned)(skiprows * CPITCH * 2);
      u32x4 g0;
      g0.x = 1u;                                           // count=1
      g0.y = lds_addr;                                     // lds_addr
      g0.z = (unsigned)(gaddr & 0xFFFFFFFFull);            // global_addr lo
      g0.w = (unsigned)((gaddr >> 32) & 0x01FFFFFFull) | (2u << 30);  // hi | type=2
      i32x8 g1;
      g1[0] = (int)((1u << 16)       // data_size = 2B
                  | (1u << 20)       // pad_enable
                  | (5u << 22)       // pad_interval: 64 dwords (=128 halves)
                  | (3u << 25));     // pad_amount: 4 dwords (=8 halves) -> CPITCH
      g1[1] = (int)(128u << 16);     // tensor_dim0[15:0] = 128 (channels)
      g1[2] = (int)((unsigned)WN_T << 16);  // tensor_dim1[15:0] = T
      g1[3] = (int)(128u << 16);     // tile_dim0 = 128 channels
      g1[4] = TT - skiprows;         // tile_dim1 = rows to move
      g1[5] = WN_R;                  // tensor_dim0_stride = 128
      g1[6] = 0; g1[7] = 0;
      const i32x4 z4 = {0, 0, 0, 0};
      const i32x8 z8 = {0, 0, 0, 0, 0, 0, 0, 0};
      __builtin_amdgcn_tensor_load_to_lds(g0, g1, z4, z4, z8, 0);
      __builtin_amdgcn_s_wait_tensorcnt(0);
    }
#else
    for (int idx = tid; idx < TT * 16; idx += 256) {
      const int j = idx >> 4, c8 = (idx & 15) << 3;
      const int t = tstart + j;
      v8bf v = zero_v8bf();
      if (t >= 0) v = *(const v8bf*)(hin_b + (size_t)t * WN_R + c8);
      *(v8bf*)(sh + j * CPITCH + c8) = v;
    }
    (void)skiprows;
#endif
    __syncthreads();

#pragma unroll
    for (int ks = 0; ks < 4; ++ks) {
      const v16bf aT = load_a_packed(pdil + kk * 32768, wave * 4 + ks);
      const v16bf aS = load_a_packed(pdil + kk * 32768, (wave + 8) * 4 + ks);
#pragma unroll
      for (int nb = 0; nb < 8; ++nb) {
        const v16bf bb = load_b_lds(sh, CPITCH, ks * 32, nb * 16);
        accT[nb] = wmma_bf16(aT, bb, accT[nb]);
        accS[nb] = wmma_bf16(aS, bb, accS[nb]);
      }
    }
  }

  __syncthreads();                           // taps done; z overwrites sh
#pragma unroll
  for (int nb = 0; nb < 8; ++nb) {
    v8bf zv;
#pragma unroll
    for (int r = 0; r < 8; ++r) {
      const float zt = tanhf(accT[nb][r]);
      const float zs = 1.0f / (1.0f + __expf(-accS[nb][r]));
      zv[r] = (__bf16)(zt * zs);
    }
    *(v8bf*)(sh + (size_t)(nb * 16 + col0) * CPITCH + wave * 16 + mrow) = zv;
  }
  __syncthreads();

  // residual GEMM (128x128 @ z): wave owns M-block 'wave'
  {
    v8f acc[8];
#pragma unroll
    for (int nb = 0; nb < 8; ++nb)
#pragma unroll
      for (int r = 0; r < 8; ++r) acc[nb][r] = res_b[wave * 16 + mrow + r];
#pragma unroll
    for (int ks = 0; ks < 4; ++ks) {
      const v16bf a = load_a_packed(pres, wave * 4 + ks);
#pragma unroll
      for (int nb = 0; nb < 8; ++nb) {
        const v16bf bb = load_b_lds(sh, CPITCH, ks * 32, nb * 16);
        acc[nb] = wmma_bf16(a, bb, acc[nb]);
      }
    }
    __bf16* hout_b = h_out + (size_t)b * WN_T * WN_R;
#pragma unroll
    for (int nb = 0; nb < 8; ++nb) {
      const size_t off = (size_t)(t0 + nb * 16 + col0) * WN_R + wave * 16 + mrow;
      const v8bf hold = *(const v8bf*)(hin_b + off);
      v8bf hv;
#pragma unroll
      for (int r = 0; r < 8; ++r) hv[r] = (__bf16)((float)hold[r] + acc[nb][r]);
      *(v8bf*)(hout_b + off) = hv;
    }
  }

  // skip GEMM (64x128 @ z): waves 0..3 own M-blocks 0..3
  if (wave < 4) {
    v8f acc[8];
#pragma unroll
    for (int nb = 0; nb < 8; ++nb)
#pragma unroll
      for (int r = 0; r < 8; ++r) acc[nb][r] = skip_b[wave * 16 + mrow + r];
#pragma unroll
    for (int ks = 0; ks < 4; ++ks) {
      const v16bf a = load_a_packed(pskip, wave * 4 + ks);
#pragma unroll
      for (int nb = 0; nb < 8; ++nb) {
        const v16bf bb = load_b_lds(sh, CPITCH, ks * 32, nb * 16);
        acc[nb] = wmma_bf16(a, bb, acc[nb]);
      }
    }
    float* sk_b = skip_acc + (size_t)b * WN_T * WN_S;
#pragma unroll
    for (int nb = 0; nb < 8; ++nb) {
      float* sp = sk_b + (size_t)(t0 + nb * 16 + col0) * WN_S + wave * 16 + mrow;
      v8f prev;
      if (first) {
#pragma unroll
        for (int r = 0; r < 8; ++r) prev[r] = 0.0f;
      } else {
        prev = *(const v8f*)sp;
      }
      *(v8f*)sp = prev + acc[nb];
    }
  }
}

// ---------------------------------------------------------------------------
// Bottom front: c_in = 1 -> outer product, vectorized stores (time-major).
// ---------------------------------------------------------------------------
__global__ void wn_front_bottom(const float* __restrict__ x,
                                const float* __restrict__ fw,
                                const float* __restrict__ fb,
                                __bf16* __restrict__ h_out)   // [B][T][R]
{
  const int i = blockIdx.x * blockDim.x + threadIdx.x;   // over B*T
  if (i >= WN_B * WN_T) return;
  const float xv = x[i];
  __bf16* hb = h_out + (size_t)i * WN_R;
#pragma unroll 4
  for (int c8 = 0; c8 < WN_R; c8 += 8) {
    v8bf v;
#pragma unroll
    for (int q = 0; q < 8; ++q) v[q] = (__bf16)(fw[c8 + q] * xv + fb[c8 + q]);
    *(v8bf*)(hb + c8) = v;
  }
}

// ---------------------------------------------------------------------------
// Head front: 128x32 GEMM over bottom output z ([B][T][32]), K=32.
// ---------------------------------------------------------------------------
__global__ __launch_bounds__(256, 1) void wn_front_head(
    const __bf16* __restrict__ z, const __bf16* __restrict__ pf,
    const float* __restrict__ fb, __bf16* __restrict__ h_out)
{
  __shared__ __align__(32) __bf16 shz[TT * ZPITCH];
  const int tid = threadIdx.x, wave = tid >> 5, lane = tid & 31;
  const int b  = blockIdx.x / NTILE;
  const int t0 = (blockIdx.x % NTILE) * TT;
  const int mrow = (lane >> 4) << 3, col0 = lane & 15;

  const __bf16* zb = z + (size_t)b * WN_T * 32;
  for (int idx = tid; idx < TT * 4; idx += 256) {
    const int j = idx >> 2, c8 = (idx & 3) << 3;
    *(v8bf*)(shz + j * ZPITCH + c8) = *(const v8bf*)(zb + (size_t)(t0 + j) * 32 + c8);
  }
  __syncthreads();

  const v16bf a = load_a_packed(pf, wave);
  __bf16* hb = h_out + (size_t)b * WN_T * WN_R;
#pragma unroll
  for (int nb = 0; nb < 8; ++nb) {
    v8f acc;
#pragma unroll
    for (int r = 0; r < 8; ++r) acc[r] = fb[wave * 16 + mrow + r];
    const v16bf bb = load_b_lds(shz, ZPITCH, 0, nb * 16);
    acc = wmma_bf16(a, bb, acc);
    v8bf hv;
#pragma unroll
    for (int r = 0; r < 8; ++r) hv[r] = (__bf16)acc[r];
    *(v8bf*)(hb + (size_t)(t0 + nb * 16 + col0) * WN_R + wave * 16 + mrow) = hv;
  }
}

// ---------------------------------------------------------------------------
// Post: out = relu(skip); y1 = relu(W1 out + b1); y2 = W2 y1 + b2.
// Bottom writes bf16 z-buffer [B][T][32]; heads write f32 into d_out.
// ---------------------------------------------------------------------------
__global__ __launch_bounds__(256, 1) void wn_post(
    const float* __restrict__ skip,          // [B][T][S]
    const __bf16* __restrict__ pe1, const float* __restrict__ b1,
    const __bf16* __restrict__ pe2, const float* __restrict__ b2,
    int out_ch, __bf16* __restrict__ out_bf, float* __restrict__ out_f32,
    long out_b_stride)
{
  __shared__ __align__(32) __bf16 sh1[TT * SPITCH];
  __shared__ __align__(32) __bf16 sh2[TT * SPITCH];
  const int tid = threadIdx.x, wave = tid >> 5, lane = tid & 31;
  const int b  = blockIdx.x / NTILE;
  const int t0 = (blockIdx.x % NTILE) * TT;
  const int mrow = (lane >> 4) << 3, col0 = lane & 15;

  const float* sk = skip + (size_t)b * WN_T * WN_S;
  for (int idx = tid; idx < TT * 8; idx += 256) {
    const int j = idx >> 3, c8 = (idx & 7) << 3;
    const v8f v = *(const v8f*)(sk + (size_t)(t0 + j) * WN_S + c8);
    v8bf o;
#pragma unroll
    for (int q = 0; q < 8; ++q) o[q] = (__bf16)(v[q] > 0.0f ? v[q] : 0.0f);
    *(v8bf*)(sh1 + j * SPITCH + c8) = o;
  }
  __syncthreads();

  // stage 1: 64x64 GEMM; wave -> M-block (wave&3), N-blocks 4*(wave>>2)..+3
  {
    const int mb = wave & 3;
    const int nbase = (wave >> 2) * 4;
    v8f acc[4];
#pragma unroll
    for (int i = 0; i < 4; ++i)
#pragma unroll
      for (int r = 0; r < 8; ++r) acc[i][r] = b1[mb * 16 + mrow + r];
#pragma unroll
    for (int ks = 0; ks < 2; ++ks) {
      const v16bf a = load_a_packed(pe1, mb * 2 + ks);
#pragma unroll
      for (int i = 0; i < 4; ++i) {
        const v16bf bb = load_b_lds(sh1, SPITCH, ks * 32, (nbase + i) * 16);
        acc[i] = wmma_bf16(a, bb, acc[i]);
      }
    }
#pragma unroll
    for (int i = 0; i < 4; ++i) {
      v8bf o;
#pragma unroll
      for (int r = 0; r < 8; ++r) {
        const float v = acc[i][r];
        o[r] = (__bf16)(v > 0.0f ? v : 0.0f);
      }
      *(v8bf*)(sh2 + (size_t)((nbase + i) * 16 + col0) * SPITCH + mb * 16 + mrow) = o;
    }
  }
  __syncthreads();

  // stage 2: out_ch x 64 GEMM
  const int ntiles = (out_ch >> 4) * 8;
  for (int tidx = wave; tidx < ntiles; tidx += 8) {
    const int mb = tidx >> 3, nb = tidx & 7;
    v8f acc;
#pragma unroll
    for (int r = 0; r < 8; ++r) acc[r] = b2[mb * 16 + mrow + r];
#pragma unroll
    for (int ks = 0; ks < 2; ++ks) {
      const v16bf a = load_a_packed(pe2, mb * 2 + ks);
      const v16bf bb = load_b_lds(sh2, SPITCH, ks * 32, nb * 16);
      acc = wmma_bf16(a, bb, acc);
    }
    if (out_bf) {                 // bottom: [B][T][32] bf16
      v8bf o;
#pragma unroll
      for (int r = 0; r < 8; ++r) o[r] = (__bf16)acc[r];
      *(v8bf*)(out_bf + (size_t)b * WN_T * out_ch +
               (size_t)(t0 + nb * 16 + col0) * out_ch + mb * 16 + mrow) = o;
    } else {                      // head: d_out channel-major [.,ch,T]
#pragma unroll
      for (int r = 0; r < 8; ++r)
        out_f32[(size_t)b * out_b_stride + (size_t)(mb * 16 + mrow + r) * WN_T +
                t0 + nb * 16 + col0] = acc[r];
    }
  }
}

// ---------------------------------------------------------------------------
// Host driver.
// ---------------------------------------------------------------------------
static void repack(const float* W, __bf16* out, int nmat, long in_s, long out_s,
                   int nmb, int nks, int row_stride, int col_stride,
                   hipStream_t st) {
  const long total = (long)nmat * nmb * nks * 512;
  wn_repack_a<<<(int)((total + 255) / 256), 256, 0, st>>>(
      W, out, nmat, in_s, out_s, nmb, nks, row_stride, col_stride);
}

extern "C" void kernel_launch(void* const* d_in, const int* in_sizes, int n_in,
                              void* d_out, int out_size, void* d_ws, size_t ws_size,
                              hipStream_t stream)
{
  (void)in_sizes; (void)n_in; (void)out_size; (void)ws_size;
  const float* x    = (const float*)d_in[0];
  const float* bfw  = (const float*)d_in[1];
  const float* bfb  = (const float*)d_in[2];
  const float* bdw  = (const float*)d_in[3];
  const float* bdb  = (const float*)d_in[4];
  const float* brw  = (const float*)d_in[5];
  const float* brb  = (const float*)d_in[6];
  const float* bsw  = (const float*)d_in[7];
  const float* bsb  = (const float*)d_in[8];
  const float* be1w = (const float*)d_in[9];
  const float* be1b = (const float*)d_in[10];
  const float* be2w = (const float*)d_in[11];
  const float* be2b = (const float*)d_in[12];
  const float* hfw  = (const float*)d_in[13];
  const float* hfb  = (const float*)d_in[14];
  const float* hdw  = (const float*)d_in[15];
  const float* hdb  = (const float*)d_in[16];
  const float* hrw  = (const float*)d_in[17];
  const float* hrb  = (const float*)d_in[18];
  const float* hsw  = (const float*)d_in[19];
  const float* hsb  = (const float*)d_in[20];
  const float* he1w = (const float*)d_in[21];
  const float* he1b = (const float*)d_in[22];
  const float* he2w = (const float*)d_in[23];
  const float* he2b = (const float*)d_in[24];

  float* out = (float*)d_out;

  // workspace layout (bf16 region first, then f32 skip)
  __bf16* p = (__bf16*)d_ws;
  __bf16* hA    = p; p += (size_t)WN_B * WN_T * WN_R;          // 4M halves
  __bf16* hB    = p; p += (size_t)WN_B * WN_T * WN_R;
  __bf16* zbuf  = p; p += (size_t)WN_B * WN_T * 32;
  __bf16* pBdil = p; p += (size_t)WN_L * 98304;                // [L][3][64][512]
  __bf16* pBres = p; p += (size_t)WN_L * 16384;
  __bf16* pBskp = p; p += (size_t)WN_L * 8192;
  __bf16* pBe1  = p; p += 4096;
  __bf16* pBe2  = p; p += 2048;
  __bf16* pHfro = p; p += (size_t)4 * 4096;
  __bf16* pHdil = p; p += (size_t)4 * WN_L * 98304;
  __bf16* pHres = p; p += (size_t)4 * WN_L * 16384;
  __bf16* pHskp = p; p += (size_t)4 * WN_L * 8192;
  __bf16* pHe1  = p; p += (size_t)4 * 4096;
  __bf16* pHe2  = p; p += (size_t)4 * 16384;
  float*  skip  = (float*)((((size_t)p + 255) / 256) * 256);   // [B][T][S]

  // ---- repack all weights into WMMA A-fragment order (bf16) ----
  for (int kk = 0; kk < 3; ++kk) {
    repack(bdw + kk, pBdil + kk * 32768, WN_L, 98304, 98304, 16, 4, 384, 3, stream);
    repack(hdw + kk, pHdil + kk * 32768, 4 * WN_L, 98304, 98304, 16, 4, 384, 3, stream);
  }
  repack(brw,  pBres, WN_L, 16384, 16384, 8, 4, 128, 1, stream);
  repack(bsw,  pBskp, WN_L,  8192,  8192, 4, 4, 128, 1, stream);
  repack(be1w, pBe1,  1, 0, 0, 4, 2, 64, 1, stream);
  repack(be2w, pBe2,  1, 0, 0, 2, 2, 64, 1, stream);
  repack(hfw,  pHfro, 4,  4096,  4096, 8, 1, 32, 1, stream);
  repack(hrw,  pHres, 4 * WN_L, 16384, 16384, 8, 4, 128, 1, stream);
  repack(hsw,  pHskp, 4 * WN_L,  8192,  8192, 4, 4, 128, 1, stream);
  repack(he1w, pHe1,  4,  4096,  4096, 4, 2, 64, 1, stream);
  repack(he2w, pHe2,  4, 16384, 16384, 16, 2, 64, 1, stream);

  const dim3 grid(WN_B * NTILE), blk(256);

  // ---- bottom wavenet ----
  wn_front_bottom<<<(WN_B * WN_T + 255) / 256, 256, 0, stream>>>(x, bfw, bfb, hA);
  __bf16* hi = hA; __bf16* ho = hB;
  for (int i = 0; i < WN_L; ++i) {
    wn_layer<<<grid, blk, 0, stream>>>(
        hi, ho, skip,
        pBdil + (size_t)i * 98304, bdb + (size_t)i * 2 * WN_R,
        pBres + (size_t)i * 16384, brb + (size_t)i * WN_R,
        pBskp + (size_t)i * 8192,  bsb + (size_t)i * WN_S,
        1 << i, i == 0 ? 1 : 0);
    __bf16* t = hi; hi = ho; ho = t;
  }
  wn_post<<<grid, blk, 0, stream>>>(skip, pBe1, be1b, pBe2, be2b,
                                    32, zbuf, (float*)nullptr, 0L);

  // ---- 4 head wavenets ----
  for (int cls = 0; cls < 4; ++cls) {
    wn_front_head<<<grid, blk, 0, stream>>>(
        zbuf, pHfro + (size_t)cls * 4096, hfb + (size_t)cls * WN_R, hA);
    hi = hA; ho = hB;
    for (int i = 0; i < WN_L; ++i) {
      const size_t li = (size_t)cls * WN_L + i;
      wn_layer<<<grid, blk, 0, stream>>>(
          hi, ho, skip,
          pHdil + li * 98304, hdb + li * 2 * WN_R,
          pHres + li * 16384, hrb + li * WN_R,
          pHskp + li * 8192,  hsb + li * WN_S,
          1 << i, i == 0 ? 1 : 0);
      __bf16* t = hi; hi = ho; ho = t;
    }
    wn_post<<<grid, blk, 0, stream>>>(
        skip,
        pHe1 + (size_t)cls * 4096,  he1b + (size_t)cls * WN_S,
        pHe2 + (size_t)cls * 16384, he2b + (size_t)cls * 256,
        256, (__bf16*)nullptr,
        out + (size_t)cls * 256 * WN_T, (long)(4L * 256 * WN_T));
  }
}
